// DoMINO_11450382811535
// MI455X (gfx1250) — matrix operand order repaired and checked by reading, not compiled, not run on hardware
//
#include <hip/hip_runtime.h>
#include <hip/hip_bf16.h>
#include <math.h>

typedef _Float16 half_t;
typedef _Float16 v16h __attribute__((ext_vector_type(16)));
typedef _Float16 v8h  __attribute__((ext_vector_type(8)));
typedef float    v8f  __attribute__((ext_vector_type(8)));

#define N_PTS   98304      // 64*48*32
#define P_NB    10
#define INF     33         // 3 + 2*5*3
#define H1      64
#define H2      32
#define OUTC    8

#define WAVES_PER_BLOCK 8
#define PTS_PER_WAVE    8
#define ROWS_PER_WAVE   (PTS_PER_WAVE * P_NB)   // 80
#define TILES_PER_WAVE  (ROWS_PER_WAVE / 16)    // 5

__device__ __forceinline__ v8f wmma16(v16h a, v16h b, v8f c) {
    return __builtin_amdgcn_wmma_f32_16x16x32_f16(
        /*neg_a=*/false, a, /*neg_b=*/false, b,
        /*c_mod=*/(short)0, c, /*reuse_a=*/false, /*reuse_b=*/false);
}

// Hardware tanh (CDNA5 V_TANH_F32, branch-free vs libm tanhf/erff).
__device__ __forceinline__ float tanh_hw(float x) {
#if __has_builtin(__builtin_amdgcn_tanhf)
    return __builtin_amdgcn_tanhf(x);
#else
    float r;
    // trailing v_nop satisfies the trans-op RAW hazard (1 op before use)
    asm("v_tanh_f32 %0, %1\n\tv_nop" : "=v"(r) : "v"(x));
    return r;
#endif
}

// GELU, tanh form (branch-free, 4 FMA + 1 hw tanh)
__device__ __forceinline__ float gelu_fast(float v) {
    const float u = 0.7978845608028654f * (v + 0.044715f * v * v * v);
    return 0.5f * v * (1.0f + tanh_hw(u));
}

// feature k of one row: [x0,x1,x2, sin(x_d * f_m) (m-major), cos(...)]
__device__ __forceinline__ float featf(int k, float c0, float c1, float c2,
                                       float f0, float f1, float f2, float f3, float f4) {
    if (k >= INF) return 0.0f;
    if (k == 0) return c0;
    if (k == 1) return c1;
    if (k == 2) return c2;
    const bool is_sin = (k < 18);
    const int  s = is_sin ? (k - 3) : (k - 18);
    const int  d = s % 3;
    const int  f = s / 3;
    const float cd = (d == 0) ? c0 : ((d == 1) ? c1 : c2);
    const float fv = (f == 0) ? f0 : (f == 1) ? f1 : (f == 2) ? f2 : (f == 3) ? f3 : f4;
    const float arg = cd * fv;
    return is_sin ? __sinf(arg) : __cosf(arg);
}

// Rebuild a 16x32 f16 A-fragment (ISA A layout) from a row-major LDS tile.
// A layout: lane holds row m=lane&15; VGPR v, half b -> k = (v<4?0:16)+(v&3)*2+b + g*8 (+32*ks)
// => halves group into two contiguous 8-half runs: [ks*32+g*8 .. +7] and [ks*32+16+g*8 .. +7]
__device__ __forceinline__ v16h loadA_lds(const half_t* __restrict__ buf,
                                          int m, int ncols, int ks, int g) {
    const v8h lo = *(const v8h*)&buf[m * ncols + ks * 32 + g * 8];
    const v8h hi = *(const v8h*)&buf[m * ncols + ks * 32 + 16 + g * 8];
    v16h a;
#pragma unroll
    for (int i = 0; i < 8; ++i) { a[i] = lo[i]; a[8 + i] = hi[i]; }
    return a;
}

__global__ __launch_bounds__(256) void domino_mlp_wmma(
    const float* __restrict__ x,   const float* __restrict__ freqs,
    const float* __restrict__ W1,  const float* __restrict__ b1,
    const float* __restrict__ W2,  const float* __restrict__ b2,
    const float* __restrict__ W3,  const float* __restrict__ b3,
    float* __restrict__ out)
{
    __shared__ __align__(16) half_t h1buf[WAVES_PER_BLOCK][16 * H1];   // 16KB
    __shared__ __align__(16) half_t h2buf[WAVES_PER_BLOCK][16 * H2];   //  8KB
    __shared__ __align__(16) float  h3buf[WAVES_PER_BLOCK][ROWS_PER_WAVE * OUTC]; // 20KB
    __shared__ __align__(16) float  mbuf [WAVES_PER_BLOCK][ROWS_PER_WAVE];        // 2.5KB

    const int lane   = threadIdx.x & 31;
    const int w      = threadIdx.x >> 5;
    const int waveId = blockIdx.x * WAVES_PER_BLOCK + w;
    const int pt_base  = waveId * PTS_PER_WAVE;
    const int row_base = pt_base * P_NB;

    const int n_lane = lane & 15;   // column (B/C/D) or row (A) index within tile
    const int g      = lane >> 4;   // lane group

    // Pull weights through L2 (global_prefetch_b8)
    __builtin_prefetch(W1, 0, 1);
    __builtin_prefetch(W2, 0, 1);
    __builtin_prefetch(W3, 0, 1);

    const float f0 = freqs[0], f1 = freqs[1], f2 = freqs[2], f3 = freqs[3], f4 = freqs[4];

    // ---- Build resident weight B-fragments (ISA B layout: n = n_lane, k = ks*32 + g*8 + (j&7) + (j>>3)*16) ----
    v16h Bw1[4][2];
#pragma unroll
    for (int t = 0; t < 4; ++t)
#pragma unroll
        for (int ks = 0; ks < 2; ++ks) {
            v16h bf;
#pragma unroll
            for (int j = 0; j < 16; ++j) {
                const int k = ks * 32 + g * 8 + (j & 7) + (j >> 3) * 16;
                bf[j] = (k < INF) ? (half_t)W1[k * H1 + (t * 16 + n_lane)] : (half_t)0.0f;
            }
            Bw1[t][ks] = bf;
        }

    v16h Bw2[2][2];
#pragma unroll
    for (int t = 0; t < 2; ++t)
#pragma unroll
        for (int ks = 0; ks < 2; ++ks) {
            v16h bf;
#pragma unroll
            for (int j = 0; j < 16; ++j) {
                const int k = ks * 32 + g * 8 + (j & 7) + (j >> 3) * 16;
                bf[j] = (half_t)W2[k * H2 + (t * 16 + n_lane)];
            }
            Bw2[t][ks] = bf;
        }

    v16h Bw3;
#pragma unroll
    for (int j = 0; j < 16; ++j) {
        const int k = g * 8 + (j & 7) + (j >> 3) * 16;   // K = 32, single k-step
        Bw3[j] = (n_lane < OUTC) ? (half_t)W3[k * OUTC + n_lane] : (half_t)0.0f;
    }

    float bias1[4];
#pragma unroll
    for (int t = 0; t < 4; ++t) bias1[t] = b1[t * 16 + n_lane];
    float bias2[2];
#pragma unroll
    for (int t = 0; t < 2; ++t) bias2[t] = b2[t * 16 + n_lane];
    const float bias3 = (n_lane < OUTC) ? b3[n_lane] : 0.0f;

    // ================= 5 row-tiles of 16 rows each =================
    for (int tile = 0; tile < TILES_PER_WAVE; ++tile) {
        const int m   = n_lane;                      // this lane's A-row
        const int row = row_base + tile * 16 + m;    // global (point,neighbor) row
        const float c0 = x[row * 3 + 0];
        const float c1 = x[row * 3 + 1];
        const float c2 = x[row * 3 + 2];
        if (g == 0)
            mbuf[w][tile * 16 + m] = (fabsf(c0) > 1e-6f) ? 1.0f : 0.0f;

        // ---- Layer 1: A built in registers from Fourier features ----
        v16h a0, a1;
#pragma unroll
        for (int v = 0; v < 8; ++v)
#pragma unroll
            for (int bb = 0; bb < 2; ++bb) {
                const int j  = 2 * v + bb;
                const int kl = (v < 4 ? 0 : 16) + (v & 3) * 2 + bb + g * 8;
                a0[j] = (half_t)featf(kl,      c0, c1, c2, f0, f1, f2, f3, f4);
                a1[j] = (half_t)featf(kl + 32, c0, c1, c2, f0, f1, f2, f3, f4);
            }

        v8f acc1[4];
#pragma unroll
        for (int t = 0; t < 4; ++t) {
            v8f c;
#pragma unroll
            for (int r = 0; r < 8; ++r) c[r] = bias1[t];
            acc1[t] = c;
        }
#pragma unroll
        for (int t = 0; t < 4; ++t) acc1[t] = wmma16(a0, Bw1[t][0], acc1[t]);
#pragma unroll
        for (int t = 0; t < 4; ++t) acc1[t] = wmma16(a1, Bw1[t][1], acc1[t]);

        // GELU; D layout (row = r + 8g, col = t*16 + n_lane) -> row-major LDS f16
#pragma unroll
        for (int t = 0; t < 4; ++t)
#pragma unroll
            for (int r = 0; r < 8; ++r)
                h1buf[w][(r + 8 * g) * H1 + t * 16 + n_lane] = (half_t)gelu_fast(acc1[t][r]);

        // ---- Layer 2: 16x64 @ 64x32 ----
        v8f acc2[2];
#pragma unroll
        for (int t = 0; t < 2; ++t) {
            v8f c;
#pragma unroll
            for (int r = 0; r < 8; ++r) c[r] = bias2[t];
            acc2[t] = c;
        }
#pragma unroll
        for (int ks = 0; ks < 2; ++ks) {
            const v16h a2 = loadA_lds(h1buf[w], m, H1, ks, g);
#pragma unroll
            for (int t = 0; t < 2; ++t) acc2[t] = wmma16(a2, Bw2[t][ks], acc2[t]);
        }
#pragma unroll
        for (int t = 0; t < 2; ++t)
#pragma unroll
            for (int r = 0; r < 8; ++r)
                h2buf[w][(r + 8 * g) * H2 + t * 16 + n_lane] = (half_t)gelu_fast(acc2[t][r]);

        // ---- Layer 3: 16x32 @ 32x8 (cols 8..15 are zero-padded) ----
        v8f acc3;
#pragma unroll
        for (int r = 0; r < 8; ++r) acc3[r] = bias3;
        const v16h a3 = loadA_lds(h2buf[w], m, H2, 0, g);
        acc3 = wmma16(a3, Bw3, acc3);

        if (n_lane < OUTC) {
#pragma unroll
            for (int r = 0; r < 8; ++r)
                h3buf[w][(tile * 16 + r + 8 * g) * OUTC + n_lane] = tanh_hw(acc3[r]);
        }
    }

    // ---- Masked neighbor reduction: 8 points x 8 channels per wave ----
    for (int o = lane; o < PTS_PER_WAVE * OUTC; o += 32) {
        const int pt = o >> 3;
        const int ch = o & 7;
        float s = 0.0f;
#pragma unroll
        for (int nb = 0; nb < P_NB; ++nb) {
            const int rr = pt * P_NB + nb;
            s += h3buf[w][rr * OUTC + ch] * mbuf[w][rr];
        }
        out[ch * N_PTS + pt_base + pt] = s;   // (b, 8, nx, ny, nz) channel-major
    }
}

extern "C" void kernel_launch(void* const* d_in, const int* in_sizes, int n_in,
                              void* d_out, int out_size, void* d_ws, size_t ws_size,
                              hipStream_t stream) {
    const float* x  = (const float*)d_in[0];
    // d_in[1] = grid: unused by the reference computation
    const float* fr = (const float*)d_in[2];
    const float* W1 = (const float*)d_in[3];
    const float* b1 = (const float*)d_in[4];
    const float* W2 = (const float*)d_in[5];
    const float* b2 = (const float*)d_in[6];
    const float* W3 = (const float*)d_in[7];
    const float* b3 = (const float*)d_in[8];
    float* out = (float*)d_out;

    const int total_waves = N_PTS / PTS_PER_WAVE;          // 12288
    const int blocks      = total_waves / WAVES_PER_BLOCK; // 1536
    domino_mlp_wmma<<<blocks, 256, 0, stream>>>(x, fr, W1, b1, W2, b2, W3, b3, out);
}
